// mHCResidualWrapper_30631706755152
// MI455X (gfx1250) — compile-verified, matrix operand order checked by reading, not run on hardware
//
#include <hip/hip_runtime.h>

typedef __bf16 bf16_t;
typedef __attribute__((ext_vector_type(16))) bf16_t v16bf;
typedef __attribute__((ext_vector_type(8)))  float  v8f;

#define CIN    256
#define COUT   256
#define HW     3136
#define WIDTH  56
#define NTOT   100352   // 32 * 3136
#define LDK    36       // padded K-stride in halfs (72B: 8B aligned, 2-way banks)

__device__ __forceinline__ unsigned short f2bf(float f) {
  unsigned u = __float_as_uint(f);
  u += 0x7FFFu + ((u >> 16) & 1u);          // round-to-nearest-even
  return (unsigned short)(u >> 16);
}
__device__ __forceinline__ float bf2f(unsigned short h) {
  return __uint_as_float(((unsigned)h) << 16);
}
__device__ __forceinline__ unsigned packbf(float f) {
  unsigned short hi = f2bf(f);
  unsigned short lo = f2bf(f - bf2f(hi));
  return ((unsigned)hi << 16) | lo;
}

// ---- kernel 1: global average pool per (b,c); optionally emit hi|lo-split x ----
__global__ void pool_kernel(const float* __restrict__ x, float* __restrict__ pooled,
                            unsigned* __restrict__ xpack) {
  const int bc = blockIdx.x;                      // b*256 + c
  const float4* p4 = (const float4*)(x + (size_t)bc * HW);
  float s = 0.f;
  if (xpack) {
    uint4* o4 = (uint4*)(xpack + (size_t)bc * HW);
    for (int i = threadIdx.x; i < HW / 4; i += 256) {
      float4 v = p4[i];
      s += v.x + v.y + v.z + v.w;
      uint4 q;
      q.x = packbf(v.x); q.y = packbf(v.y); q.z = packbf(v.z); q.w = packbf(v.w);
      o4[i] = q;
    }
  } else {
    for (int i = threadIdx.x; i < HW / 4; i += 256) {
      float4 v = p4[i];
      s += v.x + v.y + v.z + v.w;
    }
  }
  __shared__ float red[256];
  red[threadIdx.x] = s; __syncthreads();
  for (int off = 128; off > 0; off >>= 1) {
    if (threadIdx.x < off) red[threadIdx.x] += red[threadIdx.x + off];
    __syncthreads();
  }
  if (threadIdx.x == 0) pooled[bc] = red[0] * (1.0f / HW);
}

// ------------- kernel 2: weight repack -> [tap][cin][cout] hi|lo ----------
__global__ void wpack_kernel(const float* __restrict__ conv_w, unsigned* __restrict__ wp) {
  const int tc = blockIdx.x;                      // t*256 + cin
  const int t = tc >> 8, cin = tc & 255;
  const int cout = threadIdx.x;
  float w = conv_w[((size_t)cout * CIN + cin) * 9 + t];
  wp[(size_t)tc * COUT + cout] = packbf(w);
}

// ------------- kernel 3: coeffs + sinkhorn -> r, g=p*s_pre, p -------------
__global__ void coeff_kernel(const float* __restrict__ pooled,
                             const float* __restrict__ phi_w, const float* __restrict__ phi_b,
                             const float* __restrict__ rms_w, const float* __restrict__ a_pre,
                             const float* __restrict__ a_post, const float* __restrict__ a_res,
                             float* __restrict__ stats) {
  const int b = blockIdx.x, tid = threadIdx.x;
  __shared__ float red[256];
  __shared__ float coef[24];
  float pc = pooled[b * 256 + tid];
  red[tid] = pc * pc; __syncthreads();
  for (int off = 128; off > 0; off >>= 1) { if (tid < off) red[tid] += red[tid + off]; __syncthreads(); }
  const float inv_rms = rsqrtf(red[0] * (1.0f / 256.0f) + 1e-20f);
  __syncthreads();
  for (int j = 0; j < 24; ++j) {
    float wsum = 0.f;
    #pragma unroll
    for (int i = 0; i < 4; ++i)
      wsum += rms_w[i * 256 + tid] * phi_w[j * 1024 + i * 256 + tid];
    red[tid] = pc * inv_rms * wsum; __syncthreads();
    for (int off = 128; off > 0; off >>= 1) { if (tid < off) red[tid] += red[tid + off]; __syncthreads(); }
    if (tid == 0) coef[j] = red[0] + phi_b[j];
    __syncthreads();
  }
  if (tid == 0) {
    const float ap = a_pre[0], aq = a_post[0], ar = a_res[0];
    float s_pre = 0.f, hp = 0.f;
    for (int k = 0; k < 4; ++k) s_pre += 1.f / (1.f + __expf(-ap * coef[k]));
    for (int k = 0; k < 4; ++k) hp    += 1.f / (1.f + __expf(-aq * coef[4 + k]));
    const float pmean = 0.5f * hp;               // mean_n of 2*sigmoid
    float M[4][4];
    for (int i = 0; i < 4; ++i)
      for (int j = 0; j < 4; ++j) M[i][j] = __expf(ar * coef[8 + i * 4 + j]);
    for (int it = 0; it < 20; ++it) {
      for (int i = 0; i < 4; ++i) {
        float rs = M[i][0] + M[i][1] + M[i][2] + M[i][3]; float inv = 1.f / rs;
        for (int j = 0; j < 4; ++j) M[i][j] *= inv;
      }
      for (int j = 0; j < 4; ++j) {
        float cs = M[0][j] + M[1][j] + M[2][j] + M[3][j]; float inv = 1.f / cs;
        for (int i = 0; i < 4; ++i) M[i][j] *= inv;
      }
    }
    const float r = 0.25f * (M[0][0] + M[1][1] + M[2][2] + M[3][3]);
    stats[b]      = r;
    stats[32 + b] = pmean * s_pre;               // scale for conv(x)
    stats[64 + b] = pmean;                       // scale for bias
  }
}

// ------------- kernel 4: implicit-GEMM 3x3 conv via bf16x3 WMMA -----------
// out = r[b]*x + g[b]*conv3x3(x) + p[b]*bias
// Register -> double-buffered-LDS pipeline: one barrier per K-chunk; global
// loads for chunk k+1 issue before the 12 WMMAs of chunk k.
template <bool PACKED>
__global__ void __launch_bounds__(256) conv_kernel(
    const float* __restrict__ x, const unsigned* __restrict__ xpack,
    const unsigned* __restrict__ wp, const float* __restrict__ conv_b,
    const float* __restrict__ stats, float* __restrict__ out) {
  __shared__ __align__(16) unsigned short As_hi[2][64 * LDK];
  __shared__ __align__(16) unsigned short As_lo[2][64 * LDK];
  __shared__ __align__(16) unsigned short Bs_hi[2][128 * LDK];
  __shared__ __align__(16) unsigned short Bs_lo[2][128 * LDK];

  const int tid  = threadIdx.x;
  const int lane = tid & 31;
  const int wave = tid >> 5;
  const int nBase    = blockIdx.x * 128;          // flattened (b,h,w) pixel base
  const int coutBase = blockIdx.y * 64;

  const int wm = wave >> 1;                       // wave's M-subtile (0..3)
  const int wn = wave & 1;                        // wave's 64-pixel N-group
  const int halfSel = lane >> 4;                  // ISA half-wave select
  const int lrow = lane & 15;
  const int kb = halfSel * 8;                     // K-base for A/B fragments

  // staging coordinates (fixed per thread)
  const int nS = tid & 127;
  const int pS = nBase + nS;
  const int bS = pS / HW;
  const int hwS = pS - bS * HW;
  const int hS = hwS / WIDTH;
  const int wS = hwS - hS * WIDTH;
  const float*    xS  = x     + (size_t)bS * CIN * HW;
  const unsigned* xpS = PACKED ? (xpack + (size_t)bS * CIN * HW) : nullptr;
  const int mS = tid & 63;

  unsigned av[8];      // staged weights (hi|lo packed)
  unsigned bu[16];     // staged activations, packed path
  float    bf[16];     // staged activations, convert path

  auto ld = [&](int kc) {
    const int tap = kc >> 3;                      // 0..8
    const int cb  = (kc & 7) << 5;                // cin base (0..224)
    const int kh = tap / 3 - 1, kw = tap % 3 - 1;
    const int ih = hS + kh, iw = wS + kw;
    const bool valid = (ih >= 0) & (ih < WIDTH) & (iw >= 0) & (iw < WIDTH);
    #pragma unroll
    for (int e = 0; e < 8; ++e) {
      const int kl = e * 4 + (tid >> 6);
      av[e] = wp[(size_t)(tap * 256 + cb + kl) * COUT + coutBase + mS];
    }
    #pragma unroll
    for (int e = 0; e < 16; ++e) {
      const int kl = e * 2 + (tid >> 7);
      const size_t off = (size_t)(cb + kl) * HW + ih * WIDTH + iw;
      if (PACKED) bu[e] = valid ? xpS[off] : 0u;
      else        bf[e] = valid ? xS[off]  : 0.f;
    }
  };
  auto st = [&](int bufi) {
    #pragma unroll
    for (int e = 0; e < 8; ++e) {
      const int kl = e * 4 + (tid >> 6);
      As_hi[bufi][mS * LDK + kl] = (unsigned short)(av[e] >> 16);
      As_lo[bufi][mS * LDK + kl] = (unsigned short)(av[e] & 0xFFFFu);
    }
    #pragma unroll
    for (int e = 0; e < 16; ++e) {
      const int kl = e * 2 + (tid >> 7);
      const unsigned pk = PACKED ? bu[e] : packbf(bf[e]);
      Bs_hi[bufi][nS * LDK + kl] = (unsigned short)(pk >> 16);
      Bs_lo[bufi][nS * LDK + kl] = (unsigned short)(pk & 0xFFFFu);
    }
  };

  v8f acc[4] = {};

  ld(0); st(0);
  for (int kc = 0; kc < 72; ++kc) {
    __syncthreads();                              // buf[kc&1] visible to all waves
    const int cur = kc & 1;
    if (kc + 1 < 72) ld(kc + 1);                  // long-latency loads overlap WMMAs

    union Frag { v16bf v; uint2 q[4]; } ahi, alo;
    {
      const unsigned short* h = As_hi[cur] + (wm * 16 + lrow) * LDK + kb;
      const unsigned short* l = As_lo[cur] + (wm * 16 + lrow) * LDK + kb;
      ahi.q[0] = *(const uint2*)(h);      ahi.q[1] = *(const uint2*)(h + 4);
      ahi.q[2] = *(const uint2*)(h + 16); ahi.q[3] = *(const uint2*)(h + 20);
      alo.q[0] = *(const uint2*)(l);      alo.q[1] = *(const uint2*)(l + 4);
      alo.q[2] = *(const uint2*)(l + 16); alo.q[3] = *(const uint2*)(l + 20);
    }
    #pragma unroll
    for (int nj = 0; nj < 4; ++nj) {
      union Frag bhi, blo;
      const unsigned short* h = Bs_hi[cur] + (wn * 64 + nj * 16 + lrow) * LDK + kb;
      const unsigned short* l = Bs_lo[cur] + (wn * 64 + nj * 16 + lrow) * LDK + kb;
      bhi.q[0] = *(const uint2*)(h);      bhi.q[1] = *(const uint2*)(h + 4);
      bhi.q[2] = *(const uint2*)(h + 16); bhi.q[3] = *(const uint2*)(h + 20);
      blo.q[0] = *(const uint2*)(l);      blo.q[1] = *(const uint2*)(l + 4);
      blo.q[2] = *(const uint2*)(l + 16); blo.q[3] = *(const uint2*)(l + 20);
      // bf16x3 split product: hi*hi + hi*lo + lo*hi, f32 accumulate
      acc[nj] = __builtin_amdgcn_wmma_f32_16x16x32_bf16(false, ahi.v, false, bhi.v, (short)0, acc[nj], false, false);
      acc[nj] = __builtin_amdgcn_wmma_f32_16x16x32_bf16(false, ahi.v, false, blo.v, (short)0, acc[nj], false, false);
      acc[nj] = __builtin_amdgcn_wmma_f32_16x16x32_bf16(false, alo.v, false, bhi.v, (short)0, acc[nj], false, false);
    }
    if (kc + 1 < 72) st((kc + 1) & 1);            // readers of this buffer passed barrier
  }

  // epilogue: out = r*x + g*acc + p*bias   (C/D layout: VGPR j -> M=j+8*halfSel, N=lane%16)
  #pragma unroll
  for (int nj = 0; nj < 4; ++nj) {
    const int p = nBase + wn * 64 + nj * 16 + lrow;
    const int b = p / HW;
    const int hw = p - b * HW;
    const float r = stats[b], g = stats[32 + b], pm = stats[64 + b];
    #pragma unroll
    for (int j = 0; j < 8; ++j) {
      const int co = coutBase + wm * 16 + halfSel * 8 + j;
      const size_t idx = (size_t)(b * COUT + co) * HW + hw;
      out[idx] = r * x[idx] + g * acc[nj][j] + pm * conv_b[co];
    }
  }
}

extern "C" void kernel_launch(void* const* d_in, const int* in_sizes, int n_in,
                              void* d_out, int out_size, void* d_ws, size_t ws_size,
                              hipStream_t stream) {
  const float* x       = (const float*)d_in[0];
  const float* phi_w   = (const float*)d_in[1];
  const float* phi_b   = (const float*)d_in[2];
  const float* rms_w   = (const float*)d_in[3];
  const float* a_pre   = (const float*)d_in[4];
  const float* a_post  = (const float*)d_in[5];
  const float* a_res   = (const float*)d_in[6];
  const float* conv_w  = (const float*)d_in[7];
  const float* conv_b  = (const float*)d_in[8];
  float* out = (float*)d_out;

  const size_t XPACK_BYTES = (size_t)NTOT * CIN * 4;          // 102,760,448
  const size_t TAIL_BYTES  = 36864 + (size_t)9 * 256 * 256 * 4;
  const bool packed = ws_size >= XPACK_BYTES + TAIL_BYTES;

  char* ws = (char*)d_ws;
  unsigned* xpack = nullptr;
  char* tail = ws;
  if (packed) { xpack = (unsigned*)ws; tail = ws + XPACK_BYTES; }
  float*    pooled = (float*)tail;                 //  8192 floats (32 KB)
  float*    stats  = (float*)(tail + 32768);       //  96 floats
  unsigned* wp     = (unsigned*)(tail + 36864);    //  9*256*256 uints (~2.25 MB)

  pool_kernel <<<dim3(32 * 256), 256, 0, stream>>>(x, pooled, xpack);
  wpack_kernel<<<dim3(9 * 256),  256, 0, stream>>>(conv_w, wp);
  coeff_kernel<<<dim3(32),       256, 0, stream>>>(pooled, phi_w, phi_b, rms_w,
                                                   a_pre, a_post, a_res, stats);
  if (packed)
    conv_kernel<true> <<<dim3(NTOT / 128, COUT / 64), 256, 0, stream>>>(x, xpack, wp, conv_b, stats, out);
  else
    conv_kernel<false><<<dim3(NTOT / 128, COUT / 64), 256, 0, stream>>>(x, nullptr, wp, conv_b, stats, out);
}
